// ParallelPointcloudEncoder_88252987998636
// MI455X (gfx1250) — compile-verified, matrix operand order checked by reading, not run on hardware
//
#include <hip/hip_runtime.h>

// ---------------------------------------------------------------------------
// Point-cloud encoder + group self-attention for MI455X (gfx1250, wave32).
// All GEMMs via v_wmma_f32_16x16x32_f16 (f16 in, f32 accumulate).
// ---------------------------------------------------------------------------

#define DEV __device__ __forceinline__

typedef __attribute__((ext_vector_type(16))) _Float16 v16h;
typedef __attribute__((ext_vector_type(8)))  float    v8f;

#define BB      16
#define NN      8192
#define GG      512
#define MM      32
#define ENCC    384
#define NHEAD   6
#define DHEAD   64
#define INNER   384
#define EPSV    1e-5f
#define ATTSCL  0.125f   // 64^-0.5

DEV unsigned lane_u() { return threadIdx.x & 31u; }

DEV v8f zero8() {
  v8f z;
#pragma unroll
  for (int i = 0; i < 8; ++i) z[i] = 0.0f;
  return z;
}

DEV v8f wmma_f16(v16h a, v16h b, v8f c) {
  // D = A(16x32) * B(32x16) + C, f32 accumulate
  return __builtin_amdgcn_wmma_f32_16x16x32_f16(
      false, a, false, b, (short)0, c, false, false);
}

// Fragment loader for a 16x32 f16 tile whose 16 "fragment rows" are contiguous
// rows of a row-major matrix (ld halfwords per row). Works for the A operand
// (rows = M) and for the B operand when the weight is stored [out][in] = [N][K]
// (rows = N, contiguous K).  Layout per ISA 7.12.2 (16-bit A 16x32, wave32):
//   lane L: row = r0 + (L&15), half = L>>4
//   vgpr i (elements 2i,2i+1): k = k0 + (i<4?0:16) + half*8 + 2*(i&3) + {0,1}
DEV v16h load_frag_rowmajor(const _Float16* p, int ld, int r0, int k0) {
  unsigned l = lane_u();
  int hf = (int)(l >> 4);
  const _Float16* rp = p + (size_t)(r0 + (int)(l & 15u)) * ld + k0 + hf * 8;
  v16h f;
#pragma unroll
  for (int i = 0; i < 8; ++i) {
    int kk = ((i < 4) ? 0 : 16) + 2 * (i & 3);
    f[2 * i]     = rp[kk];
    f[2 * i + 1] = rp[kk + 1];
  }
  return f;
}

// A-fragment where all 16 rows are identical (broadcast of a K-vector).
DEV v16h load_frag_bcast(const _Float16* vec, int k0) {
  unsigned l = lane_u();
  int hf = (int)(l >> 4);
  const _Float16* p = vec + k0 + hf * 8;
  v16h f;
#pragma unroll
  for (int i = 0; i < 8; ++i) {
    int kk = ((i < 4) ? 0 : 16) + 2 * (i & 3);
    f[2 * i]     = p[kk];
    f[2 * i + 1] = p[kk + 1];
  }
  return f;
}

// B-fragment from a [K][N] row-major matrix (contiguous N): column c0..c0+15,
// K window [k0,k0+32). Used for the V operand in attention (P @ V).
DEV v16h load_frag_kmajor(const _Float16* p, int ld, int c0, int k0) {
  unsigned l = lane_u();
  int hf = (int)(l >> 4);
  const _Float16* cp = p + (size_t)(k0 + hf * 8) * ld + c0 + (int)(l & 15u);
  v16h f;
#pragma unroll
  for (int i = 0; i < 8; ++i) {
    int kk = ((i < 4) ? 0 : 16) + 2 * (i & 3);
    f[2 * i]     = cp[(size_t)kk * ld];
    f[2 * i + 1] = cp[(size_t)(kk + 1) * ld];
  }
  return f;
}

// ---------------------------------------------------------------------------
// Weight conversion (f32 -> f16); w1 padded from [128][6] to [128][32].
// ---------------------------------------------------------------------------
__global__ void cvt_f16_kernel(const float* __restrict__ s,
                               _Float16* __restrict__ d, int n) {
  int i = blockIdx.x * 256 + threadIdx.x;
  if (i < n) d[i] = (_Float16)s[i];
}

__global__ void cvt_w1_kernel(const float* __restrict__ s,
                              _Float16* __restrict__ d) {
  int i = blockIdx.x * 256 + threadIdx.x;  // 128*32 total
  if (i < 128 * 32) {
    int o = i >> 5, kk = i & 31;
    d[i] = (kk < 6) ? (_Float16)s[o * 6 + kk] : (_Float16)0.0f;
  }
}

// ---------------------------------------------------------------------------
// Farthest point sampling: one workgroup per batch, dist array in LDS.
// ---------------------------------------------------------------------------
__global__ __launch_bounds__(1024) void fps_kernel(
    const float* __restrict__ xyz, int* __restrict__ fidx) {
  __shared__ float dist[NN];
  __shared__ float rv[1024];
  __shared__ int   ri[1024];
  __shared__ int   s_cur;
  int b = blockIdx.x, tid = threadIdx.x;
  const float* px = xyz + (size_t)b * NN * 3;
  for (int n = tid; n < NN; n += 1024) dist[n] = 1e10f;
  if (tid == 0) s_cur = 0;
  __syncthreads();
  for (int g = 0; g < GG; ++g) {
    int cur = s_cur;
    if (tid == 0) fidx[b * GG + g] = cur;
    float cx = px[cur * 3 + 0], cy = px[cur * 3 + 1], cz = px[cur * 3 + 2];
    float bestv = -1.0f; int besti = 0;
    for (int n = tid; n < NN; n += 1024) {
      float dx = px[n * 3 + 0] - cx;
      float dy = px[n * 3 + 1] - cy;
      float dz = px[n * 3 + 2] - cz;
      float dn = fminf(dist[n], dx * dx + dy * dy + dz * dz);
      dist[n] = dn;
      if (dn > bestv) { bestv = dn; besti = n; }
    }
    rv[tid] = bestv; ri[tid] = besti;
    __syncthreads();
    for (int s = 512; s > 0; s >>= 1) {
      if (tid < s) {
        float ov = rv[tid + s]; int oi = ri[tid + s];
        if (ov > rv[tid] || (ov == rv[tid] && oi < ri[tid])) {
          rv[tid] = ov; ri[tid] = oi;
        }
      }
      __syncthreads();
    }
    if (tid == 0) s_cur = ri[0];
    __syncthreads();
  }
}

// ---------------------------------------------------------------------------
// KNN (top-32 by selection) + relative-feature build, f16 output padded K=32.
// One workgroup per (b,g).
// ---------------------------------------------------------------------------
__global__ __launch_bounds__(256) void knn_feat_kernel(
    const float* __restrict__ xyz, const float* __restrict__ color,
    const int* __restrict__ fidx, _Float16* __restrict__ feat) {
  __shared__ float dist[NN];
  __shared__ float rv[256];
  __shared__ int   ri[256];
  int bg = blockIdx.x;
  int b = bg / GG;
  int tid = threadIdx.x;
  const float* px = xyz + (size_t)b * NN * 3;
  const float* pc = color + (size_t)b * NN * 3;
  int cidx = fidx[bg];
  float cx = px[cidx * 3 + 0], cy = px[cidx * 3 + 1], cz = px[cidx * 3 + 2];
  for (int n = tid; n < NN; n += 256) {
    float dx = px[n * 3 + 0] - cx;
    float dy = px[n * 3 + 1] - cy;
    float dz = px[n * 3 + 2] - cz;
    dist[n] = dx * dx + dy * dy + dz * dz;
  }
  _Float16* fo = feat + (size_t)bg * MM * 32;
  for (int i = tid; i < MM * 32 / 2; i += 256) ((unsigned*)fo)[i] = 0u;
  __syncthreads();
  for (int m = 0; m < MM; ++m) {
    float bestv = 3.0e38f; int besti = 0;
    for (int n = tid; n < NN; n += 256) {
      float d = dist[n];
      if (d < bestv) { bestv = d; besti = n; }
    }
    rv[tid] = bestv; ri[tid] = besti;
    __syncthreads();
    for (int s = 128; s > 0; s >>= 1) {
      if (tid < s) {
        float ov = rv[tid + s]; int oi = ri[tid + s];
        if (ov < rv[tid] || (ov == rv[tid] && oi < ri[tid])) {
          rv[tid] = ov; ri[tid] = oi;
        }
      }
      __syncthreads();
    }
    int w = ri[0];
    if (tid == 0) {
      dist[w] = 3.4e38f;
      fo[m * 32 + 0] = (_Float16)(px[w * 3 + 0] - cx);
      fo[m * 32 + 1] = (_Float16)(px[w * 3 + 1] - cy);
      fo[m * 32 + 2] = (_Float16)(px[w * 3 + 2] - cz);
      fo[m * 32 + 3] = (_Float16)pc[w * 3 + 0];
      fo[m * 32 + 4] = (_Float16)pc[w * 3 + 1];
      fo[m * 32 + 5] = (_Float16)pc[w * 3 + 2];
    }
    __syncthreads();
  }
}

// ---------------------------------------------------------------------------
// Per-group encoder MLP, fully fused in LDS.  8 waves, each wave owns 16x16
// output tiles.  ~242 WMMA per wave per group.
// ---------------------------------------------------------------------------
__global__ __launch_bounds__(256) void encoder_kernel(
    const _Float16* __restrict__ feat,
    const _Float16* __restrict__ w1p, const _Float16* __restrict__ w2h,
    const _Float16* __restrict__ w3h, const _Float16* __restrict__ w4h,
    const float* __restrict__ b1,  const float* __restrict__ bn1g,
    const float* __restrict__ bn1b, const float* __restrict__ bn1m,
    const float* __restrict__ bn1v, const float* __restrict__ b2,
    const float* __restrict__ b3,  const float* __restrict__ bn2g,
    const float* __restrict__ bn2b, const float* __restrict__ bn2m,
    const float* __restrict__ bn2v, const float* __restrict__ b4,
    _Float16* __restrict__ tok) {
  __shared__ _Float16 sX[32 * 32];     // padded features
  __shared__ _Float16 sH1[32 * 128];   // after conv1+bn+relu
  __shared__ _Float16 sF[32 * 256];    // after conv2 (+bias)
  __shared__ _Float16 sFg[256];        // column max of sF
  __shared__ _Float16 sH3[32 * 512];   // after conv3+bn+relu
  __shared__ float    sCol[2 * 384];   // per-mtile column maxima of conv4

  int bg = blockIdx.x, tid = threadIdx.x;
  int wave = tid >> 5;
  unsigned l = lane_u();
  int hf = (int)(l >> 4);
  int nl = (int)(l & 15u);

  { // stage features to LDS
    const unsigned* fsrc = (const unsigned*)(feat + (size_t)bg * 1024);
    unsigned* fdst = (unsigned*)sX;
    for (int i = tid; i < 512; i += 256) fdst[i] = fsrc[i];
  }
  __syncthreads();

  // GEMM1: X[32,32] x W1p^T -> [32,128], +b1, BN1, ReLU
#pragma unroll
  for (int j = 0; j < 2; ++j) {
    int t = wave * 2 + j;
    int mt = t >> 3, nt = t & 7;
    v16h a  = load_frag_rowmajor(sX, 32, mt * 16, 0);
    v16h bf = load_frag_rowmajor(w1p, 32, nt * 16, 0);
    v8f acc = wmma_f16(a, bf, zero8());
    int col = nt * 16 + nl;
    float sc = bn1g[col] * rsqrtf(bn1v[col] + EPSV);
    float mu = bn1m[col], be = bn1b[col], bi = b1[col];
#pragma unroll
    for (int i = 0; i < 8; ++i) {
      int row = mt * 16 + i + 8 * hf;
      float v = (acc[i] + bi - mu) * sc + be;
      sH1[row * 128 + col] = (_Float16)fmaxf(v, 0.0f);
    }
  }
  __syncthreads();

  // GEMM2: [32,128] x W2^T -> [32,256], +b2
#pragma unroll
  for (int j = 0; j < 4; ++j) {
    int t = wave * 4 + j;
    int mt = t >> 4, nt = t & 15;
    v8f acc = zero8();
#pragma unroll
    for (int kt = 0; kt < 4; ++kt) {
      v16h a  = load_frag_rowmajor(sH1, 128, mt * 16, kt * 32);
      v16h bf = load_frag_rowmajor(w2h, 128, nt * 16, kt * 32);
      acc = wmma_f16(a, bf, acc);
    }
    int col = nt * 16 + nl;
    float bi = b2[col];
#pragma unroll
    for (int i = 0; i < 8; ++i) {
      int row = mt * 16 + i + 8 * hf;
      sF[row * 256 + col] = (_Float16)(acc[i] + bi);
    }
  }
  __syncthreads();

  { // global feature: column max over the 32 points
    float mv = -3.0e38f;
    for (int r = 0; r < 32; ++r) mv = fmaxf(mv, (float)sF[r * 256 + tid]);
    sFg[tid] = (_Float16)mv;
  }
  __syncthreads();

  // GEMM3: concat([fg bcast | f]) (K=512) x W3^T -> [32,512], +b3, BN2, ReLU
#pragma unroll
  for (int j = 0; j < 8; ++j) {
    int t = wave * 8 + j;
    int mt = t >> 5, nt = t & 31;
    v8f acc = zero8();
#pragma unroll
    for (int kt = 0; kt < 16; ++kt) {
      v16h a = (kt < 8) ? load_frag_bcast(sFg, kt * 32)
                        : load_frag_rowmajor(sF, 256, mt * 16, (kt - 8) * 32);
      v16h bf = load_frag_rowmajor(w3h, 512, nt * 16, kt * 32);
      acc = wmma_f16(a, bf, acc);
    }
    int col = nt * 16 + nl;
    float sc = bn2g[col] * rsqrtf(bn2v[col] + EPSV);
    float mu = bn2m[col], be = bn2b[col], bi = b3[col];
#pragma unroll
    for (int i = 0; i < 8; ++i) {
      int row = mt * 16 + i + 8 * hf;
      float v = (acc[i] + bi - mu) * sc + be;
      sH3[row * 512 + col] = (_Float16)fmaxf(v, 0.0f);
    }
  }
  __syncthreads();

  // GEMM4: [32,512] x W4^T -> [32,384]; column max over points -> token
#pragma unroll
  for (int j = 0; j < 6; ++j) {
    int t = wave * 6 + j;
    int nt = t % 24, mt = t / 24;
    v8f acc = zero8();
#pragma unroll
    for (int kt = 0; kt < 16; ++kt) {
      v16h a  = load_frag_rowmajor(sH3, 512, mt * 16, kt * 32);
      v16h bf = load_frag_rowmajor(w4h, 512, nt * 16, kt * 32);
      acc = wmma_f16(a, bf, acc);
    }
    float cm = -3.0e38f;
#pragma unroll
    for (int i = 0; i < 8; ++i) cm = fmaxf(cm, acc[i]);
    cm = fmaxf(cm, __shfl_xor(cm, 16, 32));  // combine the two 8-row halves
    if (l < 16u) sCol[mt * 384 + nt * 16 + nl] = cm;
  }
  __syncthreads();

  for (int c = tid; c < 384; c += 256) {
    float v = fmaxf(sCol[c], sCol[384 + c]) + b4[c];
    tok[(size_t)bg * 384 + c] = (_Float16)v;
  }
}

// ---------------------------------------------------------------------------
// Generic GEMM:  out[M,N] = A[M,K](f16) @ W[N,K](f16)^T + bias.
// One wave per 16x16 tile, 8 waves per block.
// ---------------------------------------------------------------------------
__global__ __launch_bounds__(256) void gemm_bias_kernel(
    const _Float16* __restrict__ A, const _Float16* __restrict__ W,
    const float* __restrict__ bias, _Float16* __restrict__ outH,
    float* __restrict__ outF, int M, int N, int K) {
  int wave = threadIdx.x >> 5;
  int ntiles = N >> 4;
  int total = (M >> 4) * ntiles;
  int tile = blockIdx.x * 8 + wave;
  if (tile >= total) return;  // wave-uniform
  int mt = tile / ntiles, nt = tile % ntiles;
  v8f acc = zero8();
  for (int k0 = 0; k0 < K; k0 += 32) {
    v16h a  = load_frag_rowmajor(A, K, mt * 16, k0);
    v16h bf = load_frag_rowmajor(W, K, nt * 16, k0);
    acc = wmma_f16(a, bf, acc);
  }
  unsigned l = lane_u();
  int hf = (int)(l >> 4);
  int n = nt * 16 + (int)(l & 15u);
  float bv = bias ? bias[n] : 0.0f;
#pragma unroll
  for (int i = 0; i < 8; ++i) {
    int row = mt * 16 + i + 8 * hf;
    float v = acc[i] + bv;
    if (outF) outF[(size_t)row * N + n] = v;
    else      outH[(size_t)row * N + n] = (_Float16)v;
  }
}

// ---------------------------------------------------------------------------
// Flash-style self-attention over G=512 tokens per (batch, head).
// Each wave owns a 16-row query tile; online softmax over 32-col score strips.
// ---------------------------------------------------------------------------
__global__ __launch_bounds__(256) void attn_kernel(
    const _Float16* __restrict__ qg, const _Float16* __restrict__ kg,
    const _Float16* __restrict__ vg, _Float16* __restrict__ og) {
  __shared__ _Float16 sP[8][16 * 32];  // per-wave P staging (C-layout -> A-layout)
  int bid = blockIdx.x;
  int chunk = bid & 3;
  int h = (bid >> 2) % NHEAD;
  int b = bid / (4 * NHEAD);
  int wave = threadIdx.x >> 5;
  unsigned l = lane_u();
  int hf = (int)(l >> 4);
  int nl = (int)(l & 15u);
  int r0 = chunk * 128 + wave * 16;   // query row in [0,512)
  size_t base = ((size_t)b * GG) * INNER + (size_t)h * DHEAD;
  const _Float16* qb = qg + base;
  const _Float16* kb = kg + base;
  const _Float16* vb = vg + base;
  _Float16* ob = og + base;

  v8f oacc[4];
#pragma unroll
  for (int t = 0; t < 4; ++t) oacc[t] = zero8();
  float rmax[8], rsum[8];
#pragma unroll
  for (int i = 0; i < 8; ++i) { rmax[i] = -3.0e38f; rsum[i] = 0.0f; }

  for (int jt = 0; jt < 16; ++jt) {
    int j0 = jt * 32;
    v8f s0 = zero8(), s1 = zero8();
#pragma unroll
    for (int dk = 0; dk < 64; dk += 32) {
      v16h a  = load_frag_rowmajor(qb, INNER, r0, dk);
      v16h k0 = load_frag_rowmajor(kb, INNER, j0, dk);
      v16h k1 = load_frag_rowmajor(kb, INNER, j0 + 16, dk);
      s0 = wmma_f16(a, k0, s0);
      s1 = wmma_f16(a, k1, s1);
    }
    _Float16* sp = sP[wave];
#pragma unroll
    for (int i = 0; i < 8; ++i) {
      float a0 = s0[i] * ATTSCL, a1 = s1[i] * ATTSCL;
      float m = fmaxf(a0, a1);
      for (int msk = 8; msk >= 1; msk >>= 1)
        m = fmaxf(m, __shfl_xor(m, msk, 16));     // per-row max (16 lanes)
      float nm = fmaxf(rmax[i], m);
      float corr = __expf(rmax[i] - nm);
      float p0 = __expf(a0 - nm), p1 = __expf(a1 - nm);
      float ps = p0 + p1;
      for (int msk = 8; msk >= 1; msk >>= 1)
        ps += __shfl_xor(ps, msk, 16);            // per-row sum
      rsum[i] = rsum[i] * corr + ps;
      rmax[i] = nm;
#pragma unroll
      for (int t = 0; t < 4; ++t) oacc[t][i] *= corr;
      int rr = i + 8 * hf;
      sp[rr * 32 + nl]      = (_Float16)p0;
      sp[rr * 32 + 16 + nl] = (_Float16)p1;
    }
    __syncthreads();
    v16h pa = load_frag_rowmajor(sp, 32, 0, 0);
#pragma unroll
    for (int t = 0; t < 4; ++t) {
      v16h vf = load_frag_kmajor(vb, INNER, t * 16, j0);
      oacc[t] = wmma_f16(pa, vf, oacc[t]);
    }
    __syncthreads();
  }

#pragma unroll
  for (int i = 0; i < 8; ++i) {
    float inv = 1.0f / rsum[i];
    int row = r0 + i + 8 * hf;
#pragma unroll
    for (int t = 0; t < 4; ++t)
      ob[(size_t)row * INNER + t * 16 + nl] = (_Float16)(oacc[t][i] * inv);
  }
}

// ---------------------------------------------------------------------------
// Host-side launch
// ---------------------------------------------------------------------------
extern "C" void kernel_launch(void* const* d_in, const int* in_sizes, int n_in,
                              void* d_out, int out_size, void* d_ws, size_t ws_size,
                              hipStream_t stream) {
  (void)in_sizes; (void)n_in; (void)out_size; (void)ws_size;
  const float* xyz   = (const float*)d_in[0];
  const float* color = (const float*)d_in[1];
  const float* w1  = (const float*)d_in[2];
  const float* b1  = (const float*)d_in[3];
  const float* bn1g = (const float*)d_in[4];
  const float* bn1b = (const float*)d_in[5];
  const float* bn1m = (const float*)d_in[6];
  const float* bn1v = (const float*)d_in[7];
  const float* w2  = (const float*)d_in[8];
  const float* b2  = (const float*)d_in[9];
  const float* w3  = (const float*)d_in[10];
  const float* b3  = (const float*)d_in[11];
  const float* bn2g = (const float*)d_in[12];
  const float* bn2b = (const float*)d_in[13];
  const float* bn2m = (const float*)d_in[14];
  const float* bn2v = (const float*)d_in[15];
  const float* w4  = (const float*)d_in[16];
  const float* b4  = (const float*)d_in[17];
  const float* wq  = (const float*)d_in[18];
  const float* wk  = (const float*)d_in[19];
  const float* wv  = (const float*)d_in[20];
  const float* wo  = (const float*)d_in[21];
  const float* bo  = (const float*)d_in[22];

  char* ws = (char*)d_ws;
  size_t off = 0;
  auto alloc = [&](size_t bytes) -> char* {
    char* p = ws + off;
    off += (bytes + 255) & ~(size_t)255;
    return p;
  };
  _Float16* w1p = (_Float16*)alloc(128 * 32 * 2);
  _Float16* w2h = (_Float16*)alloc(256 * 128 * 2);
  _Float16* w3h = (_Float16*)alloc(512 * 512 * 2);
  _Float16* w4h = (_Float16*)alloc(384 * 512 * 2);
  _Float16* wqh = (_Float16*)alloc(384 * 384 * 2);
  _Float16* wkh = (_Float16*)alloc(384 * 384 * 2);
  _Float16* wvh = (_Float16*)alloc(384 * 384 * 2);
  _Float16* woh = (_Float16*)alloc(384 * 384 * 2);
  int*      fidx = (int*)alloc((size_t)BB * GG * 4);
  _Float16* feat = (_Float16*)alloc((size_t)BB * GG * MM * 32 * 2);
  _Float16* tok  = (_Float16*)alloc((size_t)BB * GG * ENCC * 2);
  _Float16* qbuf = (_Float16*)alloc((size_t)BB * GG * INNER * 2);
  _Float16* kbuf = (_Float16*)alloc((size_t)BB * GG * INNER * 2);
  _Float16* vbuf = (_Float16*)alloc((size_t)BB * GG * INNER * 2);
  _Float16* abuf = (_Float16*)alloc((size_t)BB * GG * INNER * 2);

  // weight conversion
  cvt_w1_kernel<<<16, 256, 0, stream>>>(w1, w1p);
  cvt_f16_kernel<<<128, 256, 0, stream>>>(w2, w2h, 256 * 128);
  cvt_f16_kernel<<<1024, 256, 0, stream>>>(w3, w3h, 512 * 512);
  cvt_f16_kernel<<<768, 256, 0, stream>>>(w4, w4h, 384 * 512);
  cvt_f16_kernel<<<576, 256, 0, stream>>>(wq, wqh, 384 * 384);
  cvt_f16_kernel<<<576, 256, 0, stream>>>(wk, wkh, 384 * 384);
  cvt_f16_kernel<<<576, 256, 0, stream>>>(wv, wvh, 384 * 384);
  cvt_f16_kernel<<<576, 256, 0, stream>>>(wo, woh, 384 * 384);

  // sampling + grouping
  fps_kernel<<<BB, 1024, 0, stream>>>(xyz, fidx);
  knn_feat_kernel<<<BB * GG, 256, 0, stream>>>(xyz, color, fidx, feat);

  // per-group encoder (WMMA)
  encoder_kernel<<<BB * GG, 256, 0, stream>>>(
      feat, w1p, w2h, w3h, w4h,
      b1, bn1g, bn1b, bn1m, bn1v, b2,
      b3, bn2g, bn2b, bn2m, bn2v, b4, tok);

  // attention: QKV projections (no bias), flash core, output projection (+bo)
  const int M = BB * GG;                       // 8192 token rows
  const int gemm_blocks = (M / 16) * (INNER / 16) / 8;  // 1536
  gemm_bias_kernel<<<gemm_blocks, 256, 0, stream>>>(
      tok, wqh, nullptr, qbuf, nullptr, M, INNER, ENCC);
  gemm_bias_kernel<<<gemm_blocks, 256, 0, stream>>>(
      tok, wkh, nullptr, kbuf, nullptr, M, INNER, ENCC);
  gemm_bias_kernel<<<gemm_blocks, 256, 0, stream>>>(
      tok, wvh, nullptr, vbuf, nullptr, M, INNER, ENCC);

  attn_kernel<<<BB * NHEAD * 4, 256, 0, stream>>>(qbuf, kbuf, vbuf, abuf);

  gemm_bias_kernel<<<(M / 16) * (ENCC / 16) / 8, 256, 0, stream>>>(
      abuf, woh, bo, nullptr, (float*)d_out, M, ENCC, INNER);
}
